// Head_33474975105080
// MI455X (gfx1250) — matrix operand order repaired and checked by reading, not compiled, not run on hardware
//
#include <hip/hip_runtime.h>
#include <hip/hip_bf16.h>

typedef __attribute__((ext_vector_type(16))) _Float16 v16h;
typedef __attribute__((ext_vector_type(8)))  float    v8f;

#define Bc 4
#define Tc 4096
#define Cc 1024
#define Hc 64

__device__ __forceinline__ v8f zero8() {
    v8f z;
#pragma unroll
    for (int i = 0; i < 8; ++i) z[i] = 0.0f;
    return z;
}

__device__ __forceinline__ v16h ldg16(const _Float16* p) {
    return *(const v16h*)p;
}

// ---------------------------------------------------------------------------
// Kernel 0: convert W [C,H] fp32 -> WT [H,C] f16 (B-matrix friendly layout)
// ---------------------------------------------------------------------------
__global__ __launch_bounds__(256) void wtr_kernel(
    const float* __restrict__ wq, const float* __restrict__ wk,
    const float* __restrict__ wv,
    _Float16* __restrict__ wqT, _Float16* __restrict__ wkT,
    _Float16* __restrict__ wvT)
{
    int idx = blockIdx.x * blockDim.x + threadIdx.x;   // 0 .. 3*C*H-1
    int w   = idx / (Cc * Hc);
    int rem = idx % (Cc * Hc);
    int h   = rem / Cc;
    int c   = rem % Cc;                                // consecutive -> coalesced store
    const float* src = (w == 0) ? wq : ((w == 1) ? wk : wv);
    _Float16*    dst = (w == 0) ? wqT : ((w == 1) ? wkT : wvT);
    dst[(size_t)h * Cc + c] = (_Float16)src[(size_t)c * Hc + h];
}

// ---------------------------------------------------------------------------
// Kernel 1: q,k,v projections. One wave computes a 16x64 tile for all three
// weights, reusing a single A (x) tile per K-step. B tiles are staged 4 at a
// time in registers so the 8 global_load_b128 can clause together and WMMAs
// retire against partial loadcnt waits. Emits q,k row-major f16 and v
// transposed [B,H,T] f16.
// ---------------------------------------------------------------------------
__global__ __launch_bounds__(256) void proj_kernel(
    const float* __restrict__ x,
    const _Float16* __restrict__ wqT, const _Float16* __restrict__ wkT,
    const _Float16* __restrict__ wvT,
    _Float16* __restrict__ qf, _Float16* __restrict__ kf,
    _Float16* __restrict__ vT)
{
    const int lane = threadIdx.x & 31;
    const int wave = threadIdx.x >> 5;
    const int tile = blockIdx.x * 8 + wave;     // 0 .. B*T/16 - 1
    const int rowbase = tile * 16;
    const int l15 = lane & 15;
    const int g   = lane >> 4;

    v8f acc[3][4];
#pragma unroll
    for (int w = 0; w < 3; ++w)
#pragma unroll
        for (int n = 0; n < 4; ++n) acc[w][n] = zero8();

    const _Float16* wT[3] = {wqT, wkT, wvT};
    const float* xrow = x + (size_t)(rowbase + l15) * Cc;

    for (int k0 = 0; k0 < Cc; k0 += 32) {
        // A tile 16x32 f16: lanes 0-15 row M=lane, halves0-15 K=0..15
        //                   lanes 16-31 shift K by 16.
        v16h a;
        const float* pa = xrow + k0 + g * 16;
#pragma unroll
        for (int i = 0; i < 8; ++i) {
            a[i]     = (_Float16)pa[i];
            a[8 + i] = (_Float16)pa[8 + i];
        }
#pragma unroll
        for (int w = 0; w < 3; ++w) {
            // Stage all 4 B tiles (32 VGPRs) so loads clause + waits are partial
            v16h bt[4];
#pragma unroll
            for (int n = 0; n < 4; ++n) {
                // B tile 32x16: lane N = 16n+l15 (= h), halves = 16 consecutive K
                const _Float16* pb = wT[w] + (size_t)(16 * n + l15) * Cc + k0 + 16 * g;
                bt[n] = ldg16(pb);
            }
#pragma unroll
            for (int n = 0; n < 4; ++n)
                acc[w][n] = __builtin_amdgcn_wmma_f32_16x16x32_f16(
                    false, a, false, bt[n], (short)0, acc[w][n], false, false);
        }
    }

    const int t    = rowbase % Tc;    // row within batch (tiles never straddle batches)
    const int bidx = rowbase / Tc;
#pragma unroll
    for (int n = 0; n < 4; ++n) {
#pragma unroll
        for (int r = 0; r < 8; ++r) {
            int row = rowbase + r + 8 * g;
            int col = 16 * n + l15;
            qf[(size_t)row * Hc + col] = (_Float16)acc[0][n][r];
            kf[(size_t)row * Hc + col] = (_Float16)acc[1][n][r];
            int tt = t + r + 8 * g;
            vT[((size_t)bidx * Hc + col) * Tc + tt] = (_Float16)acc[2][n][r];
        }
    }
}

// ---------------------------------------------------------------------------
// Kernel 2: causal flash attention, software-pipelined. One wave owns 16
// queries; 32 keys/step. K tiles for step i+1 and V tiles for step i are
// issued before the softmax VALU block so global-load latency overlaps the
// ~200-instruction online-softmax + LDS repack.
// ---------------------------------------------------------------------------
__global__ __launch_bounds__(128) void attn_kernel(
    const _Float16* __restrict__ qf, const _Float16* __restrict__ kf,
    const _Float16* __restrict__ vT, float* __restrict__ out)
{
    __shared__ _Float16 pbuf[4][16 * 32];      // per-wave P staging (C/D -> A repack)
    const int lane = threadIdx.x & 31;
    const int wave = threadIdx.x >> 5;
    const int l15  = lane & 15;
    const int g    = lane >> 4;
    const int tilesPerBatch = Tc / 64;
    const int b     = blockIdx.x / tilesPerBatch;
    const int qbase = (blockIdx.x % tilesPerBatch) * 64 + wave * 16;  // within batch
    const size_t rowoff = (size_t)b * Tc;

    // Persistent A tiles of q: 16 queries x 64 h as two K=32 chunks
    v16h aq[2];
    {
        const _Float16* qrow = qf + (rowoff + qbase + l15) * Hc;
#pragma unroll
        for (int c = 0; c < 2; ++c) {
            const _Float16* p0 = qrow + 32 * c + g * 16;
#pragma unroll
            for (int i = 0; i < 8; ++i) {
                aq[c][i]     = p0[i];
                aq[c][8 + i] = p0[8 + i];
            }
        }
    }

    v8f acc[4];
#pragma unroll
    for (int n = 0; n < 4; ++n) acc[n] = zero8();
    float m[8], l[8];
#pragma unroll
    for (int r = 0; r < 8; ++r) { m[r] = -3.0e38f; l[r] = 0.0f; }

    const float scale = 0.03125f;              // C^-0.5 = 1/32
    _Float16* pl = &pbuf[wave][0];
    const int kv_last = qbase + 15;            // uniform per wave -> EXEC stays full

    // Per-lane invariant pieces of the tile addresses
    const _Float16* kbase = kf + (rowoff + l15) * Hc + 16 * g;          // + key*Hc + 32c
    const _Float16* vbase = vT + ((size_t)b * Hc + l15) * Tc + 16 * g;  // + 16n*Tc + kv0

    // Prologue: preload K B-tiles for kv0 = 0
    v16h bk[4];
#pragma unroll
    for (int j = 0; j < 2; ++j)
#pragma unroll
        for (int c = 0; c < 2; ++c)
            bk[2 * j + c] = ldg16(kbase + (size_t)(16 * j) * Hc + 32 * c);

    for (int kv0 = 0; kv0 <= kv_last; kv0 += 32) {
        // ---- S = q @ k^T for 32 keys (two 16-key subtiles) ----
        v8f s[2];
#pragma unroll
        for (int j = 0; j < 2; ++j) {
            v8f st = zero8();
#pragma unroll
            for (int c = 0; c < 2; ++c)
                st = __builtin_amdgcn_wmma_f32_16x16x32_f16(
                    false, aq[c], false, bk[2 * j + c], (short)0, st, false, false);
            int kvb = kv0 + 16 * j;
#pragma unroll
            for (int r = 0; r < 8; ++r) {
                int row = qbase + r + 8 * g;   // query index
                int col = kvb + l15;           // key index
                float v = st[r] * scale;
                st[r] = (col <= row) ? v : -3.0e38f;
            }
            s[j] = st;
        }

        // ---- issue V tiles (this step) + K tiles (next step, clamped) NOW ----
        v16h bv[4];
#pragma unroll
        for (int n = 0; n < 4; ++n)
            bv[n] = ldg16(vbase + (size_t)(16 * n) * Tc + kv0);
        int kvn = (kv0 + 32 <= kv_last) ? (kv0 + 32) : kv0;  // uniform, branchless
#pragma unroll
        for (int j = 0; j < 2; ++j)
#pragma unroll
            for (int c = 0; c < 2; ++c)
                bk[2 * j + c] = ldg16(kbase + (size_t)(kvn + 16 * j) * Hc + 32 * c);

        // ---- online softmax over the 32-key block (hides load latency) ----
#pragma unroll
        for (int r = 0; r < 8; ++r) {
            float mt = fmaxf(s[0][r], s[1][r]);
#pragma unroll
            for (int mask = 8; mask >= 1; mask >>= 1)
                mt = fmaxf(mt, __shfl_xor(mt, mask, 32));
            float mn    = fmaxf(m[r], mt);
            float alpha = __expf(m[r] - mn);
            float p0    = __expf(s[0][r] - mn);
            float p1    = __expf(s[1][r] - mn);
            float rs    = p0 + p1;
#pragma unroll
            for (int mask = 8; mask >= 1; mask >>= 1)
                rs += __shfl_xor(rs, mask, 32);
            l[r] = l[r] * alpha + rs;
            m[r] = mn;
#pragma unroll
            for (int n = 0; n < 4; ++n) acc[n][r] *= alpha;
            int row = r + 8 * g;
            pl[row * 32 + l15]      = (_Float16)p0;   // row-major P in LDS
            pl[row * 32 + 16 + l15] = (_Float16)p1;
        }
        // ---- reload P in A-matrix layout (in-wave LDS ordering via DScnt) ----
        v16h ap;
        {
            const _Float16* pr = pl + l15 * 32 + g * 16;
#pragma unroll
            for (int i = 0; i < 8; ++i) {
                ap[i]     = pr[i];
                ap[8 + i] = pr[8 + i];
            }
        }
        // ---- acc += P @ V ----
#pragma unroll
        for (int n = 0; n < 4; ++n)
            acc[n] = __builtin_amdgcn_wmma_f32_16x16x32_f16(
                false, ap, false, bv[n], (short)0, acc[n], false, false);
    }

    // ---- epilogue: normalize and store fp32 [B,T,H] ----
#pragma unroll
    for (int n = 0; n < 4; ++n) {
#pragma unroll
        for (int r = 0; r < 8; ++r) {
            int row = qbase + r + 8 * g;
            out[(rowoff + row) * Hc + 16 * n + l15] = acc[n][r] / l[r];
        }
    }
}

// ---------------------------------------------------------------------------
extern "C" void kernel_launch(void* const* d_in, const int* in_sizes, int n_in,
                              void* d_out, int out_size, void* d_ws, size_t ws_size,
                              hipStream_t stream) {
    (void)in_sizes; (void)n_in; (void)out_size; (void)ws_size;
    const float* x  = (const float*)d_in[0];
    const float* Wq = (const float*)d_in[1];
    const float* Wk = (const float*)d_in[2];
    const float* Wv = (const float*)d_in[3];
    float* out = (float*)d_out;

    char* ws = (char*)d_ws;
    _Float16* wqT = (_Float16*)(ws);                                  // 128 KB
    _Float16* wkT = (_Float16*)(ws + 128 * 1024);                     // 128 KB
    _Float16* wvT = (_Float16*)(ws + 256 * 1024);                     // 128 KB
    _Float16* qf  = (_Float16*)(ws + 512 * 1024);                     // 2 MB
    _Float16* kf  = (_Float16*)(ws + 512 * 1024 + 2 * 1024 * 1024);   // 2 MB
    _Float16* vT  = (_Float16*)(ws + 512 * 1024 + 4 * 1024 * 1024);   // 2 MB

    // 3*C*H elements / 256 threads
    wtr_kernel<<<dim3((3 * Cc * Hc) / 256), dim3(256), 0, stream>>>(
        Wq, Wk, Wv, wqT, wkT, wvT);
    // B*T/16 = 1024 row tiles, 8 waves per block
    proj_kernel<<<dim3((Bc * Tc / 16) / 8), dim3(256), 0, stream>>>(
        x, wqT, wkT, wvT, qf, kf, vT);
    // B * T/64 blocks, 4 waves (16 queries each) per block
    attn_kernel<<<dim3(Bc * (Tc / 64)), dim3(128), 0, stream>>>(
        qf, kf, vT, out);
}
